// crf_layer_43928925503828
// MI455X (gfx1250) — matrix operand order repaired and checked by reading, not compiled, not run on hardware
//
#include <hip/hip_runtime.h>

#define CDIM 64
#define NB   16
#define GS   68   // g tile row stride (dwords): 4b bank offset per batch row
#define US   36   // u tile row stride (dwords): conflict-free b128 fragment loads

typedef __attribute__((ext_vector_type(16))) __bf16 v16bf;
typedef __attribute__((ext_vector_type(8)))  float  v8f;
typedef __attribute__((ext_vector_type(4)))  unsigned int v4u;

union Frag16 {
  v16bf v;
  unsigned short h[16];
  unsigned int   u[8];
  v4u            q[2];
};

__device__ __forceinline__ unsigned int f32_to_bf16_rne(float x) {
  unsigned int u = __float_as_uint(x);
  unsigned int r = (u + 0x7FFFu + ((u >> 16) & 1u)) >> 16;
  return r & 0xFFFFu;
}
__device__ __forceinline__ float bf16_to_f32(unsigned int h) {
  return __uint_as_float(h << 16);
}

// ---------------------------------------------------------------------------
// Gold-path score: gold[b] = init[tags0] + logits[b,0,tags0]
//                  + sum_{t=1..len-1} trans[tags_t, tags_{t-1}] + logits[b,t,tags_t]
// ---------------------------------------------------------------------------
__global__ __launch_bounds__(256) void crf_gold_kernel(
    const float* __restrict__ logits, const float* __restrict__ trans,
    const float* __restrict__ init_a, const int* __restrict__ tags,
    const int* __restrict__ lengths, float* __restrict__ gold, int T)
{
  const int b = blockIdx.x;
  const int len = lengths[b];
  const int* tg = tags + (size_t)b * T;
  const float* lg = logits + (size_t)b * T * CDIM;

  float s = 0.f;
  for (int t = 1 + (int)threadIdx.x; t < len; t += 256) {
    int cur = tg[t], prev = tg[t - 1];
    s += trans[cur * CDIM + prev] + lg[(size_t)t * CDIM + cur];
  }
  __shared__ float red[256];
  red[threadIdx.x] = s;
  __syncthreads();
  for (int off = 128; off > 0; off >>= 1) {
    if ((int)threadIdx.x < off) red[threadIdx.x] += red[threadIdx.x + off];
    __syncthreads();
  }
  if (threadIdx.x == 0) {
    int t0 = tg[0];
    gold[b] = red[0] + init_a[t0] + lg[t0];
  }
}

// ---------------------------------------------------------------------------
// Main scan: one 4-wave workgroup per 16-batch tile; wave w owns M-tile w.
// State u[j][b] = exp(alpha[j]-M[b]) kept in LDS as bf16 hi+lo (batch-major,
// padded stride -> all fragment traffic is conflict-free ds_load/store_b128).
// Each step: s = E @ U via 6x V_WMMA_F32_16X16X32_BF16 (error-compensated
// bf16 split), then rescale with one log per batch:  M += gmax + log(smax),
// u' = exp(g-gmax) * s/smax.  logZ captured at t == len-1.
// ---------------------------------------------------------------------------
__global__ __launch_bounds__(128) void crf_logz_wmma(
    const float* __restrict__ logits,   // [N][T][64]
    const float* __restrict__ trans,    // [64][64]
    const float* __restrict__ init_a,   // [64]
    const int*   __restrict__ lengths,  // [N]
    const float* __restrict__ gold,     // [N] (from ws)
    float*       __restrict__ out,      // [N]
    int T)
{
  __shared__ float        g_s[NB * GS];       // logits tile, [b][i], padded
  __shared__ unsigned int uhi_s[NB * US];     // u hi bf16 pairs, [b][i/2], padded
  __shared__ unsigned int ulo_s[NB * US];     // u lo residual bf16 pairs
  __shared__ float pmax_s[128], psmax_s[128], psum_s[128];
  __shared__ float gmax_s[NB], smax_s[NB], Moff_s[NB], res_s[NB];

  const int tid   = threadIdx.x;
  const int wave  = tid >> 5;     // 0..3 : M-tile owner
  const int lane  = tid & 31;
  const int lhalf = lane & 15;
  const int hsel  = lane >> 4;
  const int b0    = blockIdx.x * NB;

  // ---- resident A fragments: E = exp(trans) split into bf16 hi + lo ----
  // A fragment, contiguous k order (B matches): lane<16 -> M=lane, K=0..15; lane>=16 -> K=16..31
  Frag16 Ehi[2], Elo[2];
  {
    const int mrow = wave * 16 + lhalf;
    for (int c = 0; c < 2; ++c) {
      for (int h = 0; h < 16; ++h) {
        int kk = c * 32 + hsel * 16 + h;
        float e = __expf(trans[mrow * CDIM + kk]);
        unsigned int hb = f32_to_bf16_rne(e);
        unsigned int lb = f32_to_bf16_rne(e - bf16_to_f32(hb));
        Ehi[c].h[h] = (unsigned short)hb;
        Elo[c].h[h] = (unsigned short)lb;
      }
    }
  }

  // cooperative logits load mapping: 8 threads per batch, 8 contiguous f32 each
  const int gb  = tid >> 3;
  const int gi0 = (tid & 7) * 8;
  const float* gbase = logits + (size_t)(b0 + gb) * T * CDIM + gi0;

  int lenb = 0;
  if (tid < NB) lenb = lengths[b0 + tid];

  // C-fragment ownership: VGPR r of lane -> (i = 16*wave + 8*hsel + r, b = lane%16)
  const int bown  = lhalf;
  const int iown0 = wave * 16 + hsel * 8;
  const int pr0   = iown0 >> 1;  // packed pair base (multiple of 4)

  // ---------------- init: t = 0 ----------------
  {
    float gv[8];
    #pragma unroll
    for (int k = 0; k < 8; ++k) gv[k] = gbase[k] + init_a[gi0 + k];
    float pm = gv[0];
    #pragma unroll
    for (int k = 1; k < 8; ++k) pm = fmaxf(pm, gv[k]);
    #pragma unroll
    for (int k = 0; k < 8; ++k) g_s[gb * GS + gi0 + k] = gv[k];
    pmax_s[tid] = pm;
    __syncthreads();
    if (tid < NB) {
      float m = pmax_s[tid * 8];
      #pragma unroll
      for (int j = 1; j < 8; ++j) m = fmaxf(m, pmax_s[tid * 8 + j]);
      gmax_s[tid] = m;
      Moff_s[tid] = m;
    }
    __syncthreads();
    float gm = gmax_s[bown];
    v4u ph, pl;
    #pragma unroll
    for (int q = 0; q < 4; ++q) {
      float u0 = __expf(g_s[bown * GS + iown0 + 2 * q    ] - gm);
      float u1 = __expf(g_s[bown * GS + iown0 + 2 * q + 1] - gm);
      unsigned int h0 = f32_to_bf16_rne(u0), h1 = f32_to_bf16_rne(u1);
      unsigned int l0 = f32_to_bf16_rne(u0 - bf16_to_f32(h0));
      unsigned int l1 = f32_to_bf16_rne(u1 - bf16_to_f32(h1));
      ph[q] = h0 | (h1 << 16);
      pl[q] = l0 | (l1 << 16);
    }
    *(__attribute__((address_space(3))) v4u*)&uhi_s[bown * US + pr0] = ph;
    *(__attribute__((address_space(3))) v4u*)&ulo_s[bown * US + pr0] = pl;
    __syncthreads();
  }

  // ---------------- scan t = 1 .. T-1 ----------------
  for (int t = 1; t < T; ++t) {
    const float* gp = gbase + (size_t)t * CDIM;
    if (t + 1 < T) __builtin_prefetch(gp + CDIM, 0, 1);   // global_prefetch_b8
    float4 ga  = *(const float4*)(gp);
    float4 gbv = *(const float4*)(gp + 4);

    // ---- GEMM: s = E @ U, error-compensated bf16 split, f32 accumulate ----
    v8f acc = (v8f){0.f, 0.f, 0.f, 0.f, 0.f, 0.f, 0.f, 0.f};
    #pragma unroll
    for (int c = 0; c < 2; ++c) {
      // B fragment, contiguous k order as A: lane<16 -> N=lane, K 0..15 (pairs/VGPR); lane>=16 -> K 16..31
      Frag16 Bhi, Blo;
      const int kb = lhalf * US + c * 16 + hsel * 8;   // batch-major, b128-aligned
      Bhi.q[0] = *(const v4u*)&uhi_s[kb];
      Bhi.q[1] = *(const v4u*)&uhi_s[kb + 4];
      Blo.q[0] = *(const v4u*)&ulo_s[kb];
      Blo.q[1] = *(const v4u*)&ulo_s[kb + 4];
      acc = __builtin_amdgcn_wmma_f32_16x16x32_bf16(false, Ehi[c].v, false, Bhi.v,
                                                    (short)0, acc, false, false);
      acc = __builtin_amdgcn_wmma_f32_16x16x32_bf16(false, Ehi[c].v, false, Blo.v,
                                                    (short)0, acc, false, false);
      acc = __builtin_amdgcn_wmma_f32_16x16x32_bf16(false, Elo[c].v, false, Bhi.v,
                                                    (short)0, acc, false, false);
    }

    // partial smax over the 8 owned s-values
    float sm = acc[0];
    #pragma unroll
    for (int r = 1; r < 8; ++r) sm = fmaxf(sm, acc[r]);
    psmax_s[tid] = sm;

    // stash logits (straight b128 stores) + partial gmax
    float pm = fmaxf(fmaxf(fmaxf(ga.x, ga.y), fmaxf(ga.z, ga.w)),
                     fmaxf(fmaxf(gbv.x, gbv.y), fmaxf(gbv.z, gbv.w)));
    pmax_s[tid] = pm;
    *(float4*)&g_s[gb * GS + gi0]     = ga;
    *(float4*)&g_s[gb * GS + gi0 + 4] = gbv;
    __syncthreads();                                  // barrier 1

    if (tid < NB) {
      float m = pmax_s[tid * 8];
      #pragma unroll
      for (int j = 1; j < 8; ++j) m = fmaxf(m, pmax_s[tid * 8 + j]);
      float s = psmax_s[tid];
      #pragma unroll
      for (int j = 1; j < 8; ++j) s = fmaxf(s, psmax_s[tid + 16 * j]);
      gmax_s[tid] = m;
      smax_s[tid] = s;
      Moff_s[tid] += m + __logf(s);                   // M_new
    }
    __syncthreads();                                  // barrier 2

    {
      float gm  = gmax_s[bown];
      float inv = 1.0f / smax_s[bown];
      float4 r0 = *(const float4*)&g_s[bown * GS + iown0];
      float4 r1 = *(const float4*)&g_s[bown * GS + iown0 + 4];
      float gr[8] = {r0.x, r0.y, r0.z, r0.w, r1.x, r1.y, r1.z, r1.w};
      float lsum = 0.f;
      v4u ph, pl;
      #pragma unroll
      for (int q = 0; q < 4; ++q) {
        float u0 = __expf(gr[2 * q]     - gm) * (acc[2 * q]     * inv);
        float u1 = __expf(gr[2 * q + 1] - gm) * (acc[2 * q + 1] * inv);
        lsum += u0 + u1;
        unsigned int h0 = f32_to_bf16_rne(u0), h1 = f32_to_bf16_rne(u1);
        unsigned int l0 = f32_to_bf16_rne(u0 - bf16_to_f32(h0));
        unsigned int l1 = f32_to_bf16_rne(u1 - bf16_to_f32(h1));
        ph[q] = h0 | (h1 << 16);
        pl[q] = l0 | (l1 << 16);
      }
      *(v4u*)&uhi_s[bown * US + pr0] = ph;            // one ds_store_b128 each
      *(v4u*)&ulo_s[bown * US + pr0] = pl;
      psum_s[tid] = lsum;
    }
    __syncthreads();                                  // barrier 3

    if (tid < NB) {
      float su = psum_s[tid];
      #pragma unroll
      for (int j = 1; j < 8; ++j) su += psum_s[tid + 16 * j];
      if (t == lenb - 1) res_s[tid] = Moff_s[tid] + __logf(su);  // logZ(len)
    }
    // next iteration's LDS writes are fenced by barriers 1/2 above
  }

  __syncthreads();
  if (tid < NB) out[b0 + tid] = res_s[tid] - gold[b0 + tid];
}

// ---------------------------------------------------------------------------
extern "C" void kernel_launch(void* const* d_in, const int* in_sizes, int n_in,
                              void* d_out, int out_size, void* d_ws, size_t ws_size,
                              hipStream_t stream) {
  const float* logits = (const float*)d_in[0];   // [N,T,64] f32
  const float* trans  = (const float*)d_in[1];   // [64,64]  f32
  const float* init_a = (const float*)d_in[2];   // [64]     f32
  const int*   lens   = (const int*)d_in[3];     // [N]      i32
  const int*   tags   = (const int*)d_in[4];     // [N,T]    i32
  float* out = (float*)d_out;

  const int N = in_sizes[3];
  const int T = in_sizes[4] / N;

  float* gold = (float*)d_ws;                    // N floats of scratch

  crf_gold_kernel<<<N, 256, 0, stream>>>(logits, trans, init_a, tags, lens, gold, T);
  crf_logz_wmma<<<N / NB, 128, 0, stream>>>(logits, trans, init_a, lens, gold, out, T);
}
